// Generator_29970281791983
// MI455X (gfx1250) — matrix operand order repaired and checked by reading, not compiled, not run on hardware
//
#include <hip/hip_runtime.h>
#include <stdint.h>

// Problem constants (match reference)
#define V_  32000
#define E_  256
#define H_  512
#define LAT_ 100
#define NL_ 3
#define B_  128
#define S_  32
#define G4H 2048   // 4*H

typedef __attribute__((ext_vector_type(16))) __bf16 v16bf;
typedef __attribute__((ext_vector_type(8)))  float  v8f;

union BF16x16 { v16bf v; unsigned short s[16]; uint4 q[2]; };

__device__ __forceinline__ unsigned short f2bf(float f) {
  // round-to-nearest-even fp32 -> bf16 (used only in one-time converts + pointwise)
  unsigned int u = __float_as_uint(f);
  unsigned int r = u + 0x7FFFu + ((u >> 16) & 1u);
  return (unsigned short)(r >> 16);
}

// A-fragment (16x32 bf16) from a bf16 row: elements 0..7 -> K=kc+half*8..+7,
// elements 8..15 -> K=kc+16+half*8..+7.  Two 16B loads, no conversion.
__device__ __forceinline__ v16bf load_a_bf(const unsigned short* __restrict__ row,
                                           int kc, int half) {
  const unsigned short* p = row + kc + half * 8;
  BF16x16 a;
  a.q[0] = *(const uint4*)p;
  a.q[1] = *(const uint4*)(p + 16);
  return a.v;
}

// B-fragment (32x16 bf16): lane r = column n = W row; elements 0..7 -> K=kc+half*8..+7, 8..15 -> K=kc+16+half*8..+7.
// two runs of 8 bf16, 16 apart = two 16B loads, no conversion.
__device__ __forceinline__ v16bf load_b_bf(const unsigned short* __restrict__ wrow,
                                           int kc, int half) {
  const unsigned short* p = wrow + kc + half * 8;
  BF16x16 b;
  b.q[0] = *(const uint4*)p;
  b.q[1] = *(const uint4*)(p + 16);
  return b.v;
}

__device__ __forceinline__ unsigned int ordf(float f) {
  unsigned int u = __float_as_uint(f);
  return (u & 0x80000000u) ? ~u : (u | 0x80000000u);
}

__device__ __forceinline__ unsigned long long shfl_xor_u64(unsigned long long v, int m) {
  unsigned int lo = (unsigned int)v, hi = (unsigned int)(v >> 32);
  lo = __shfl_xor(lo, m, 32);
  hi = __shfl_xor(hi, m, 32);
  return ((unsigned long long)hi << 32) | lo;
}

__device__ __forceinline__ float sigm(float x) { return 1.0f / (1.0f + __expf(-x)); }

// ---------------------------------------------------------------------------
// One-time fp32 -> bf16 bulk convert (2 elements per thread, packed store)
__global__ void cvt_bf16_kernel(const float* __restrict__ src,
                                unsigned int* __restrict__ dst, int n2) {
  int i = blockIdx.x * blockDim.x + threadIdx.x;
  if (i >= n2) return;
  float2 v = *(const float2*)(src + 2 * (size_t)i);
  dst[i] = (unsigned int)f2bf(v.x) | ((unsigned int)f2bf(v.y) << 16);
}

// ---------------------------------------------------------------------------
// h0 = tanh(noise @ W_lat^T + b_lat); torch .view scramble == flat reinterpret.
// Writes bf16 h-state (GEMM operand) and fp32 c-state (zeros).
__global__ void init_h0_kernel(const float* __restrict__ noise,
                               const float* __restrict__ Wlat,
                               const float* __restrict__ blat,
                               unsigned short* __restrict__ hbf,
                               float* __restrict__ cA) {
  int idx = blockIdx.x * blockDim.x + threadIdx.x;   // NL*B*H = 196608
  if (idx >= NL_ * B_ * H_) return;
  int b = idx / (NL_ * H_);
  int j = idx % (NL_ * H_);
  float acc = blat[j];
  const float* nr = noise + (size_t)b * LAT_;
  const float* wr = Wlat + (size_t)j * LAT_;
#pragma unroll 4
  for (int k = 0; k < LAT_; ++k) acc = fmaf(nr[k], wr[k], acc);
  hbf[idx] = f2bf(tanhf(acc));
  cA[idx] = 0.0f;
}

__global__ void init_misc_kernel(int* __restrict__ tok,
                                 unsigned long long* __restrict__ packed) {
  int b = threadIdx.x;
  tok[b] = 1;            // start token
  packed[b] = 0ull;
}

// ---------------------------------------------------------------------------
// Gates GEMM: G[128 x 2048] = X @ Wih^T + Hs @ Whh^T + bih + bhh   (bf16 WMMA)
// wave tile = 16x64 ; block = 4 waves ; grid = (8, 8) -> 256 waves
__global__ void __launch_bounds__(128)
gates_gemm_kernel(const unsigned short* __restrict__ Xb, const int* __restrict__ tok,
                  const unsigned short* __restrict__ Hb,
                  const unsigned short* __restrict__ Wihb,
                  const unsigned short* __restrict__ Whhb,
                  const float* __restrict__ bih, const float* __restrict__ bhh,
                  float* __restrict__ G, int Kx) {
  int wave = threadIdx.x >> 5;
  int lane = threadIdx.x & 31;
  int half = lane >> 4, r = lane & 15;
  int tileN = (blockIdx.x * 4 + wave) * 64;
  int tileM = blockIdx.y * 16;
  int arow = tileM + r;

  const unsigned short* xrow =
      tok ? (Xb + (size_t)tok[arow] * Kx) : (Xb + (size_t)arow * Kx);
  const unsigned short* hrow = Hb + (size_t)arow * H_;

  v8f acc[4] = {v8f{}, v8f{}, v8f{}, v8f{}};

  for (int kc = 0; kc < Kx; kc += 32) {
    v16bf a = load_a_bf(xrow, kc, half);
#pragma unroll
    for (int s = 0; s < 4; ++s) {
      v16bf b = load_b_bf(Wihb + (size_t)(tileN + s * 16 + r) * Kx, kc, half);
      acc[s] = __builtin_amdgcn_wmma_f32_16x16x32_bf16(false, a, false, b,
                                                       (short)0, acc[s], false, false);
    }
  }
  for (int kc = 0; kc < H_; kc += 32) {
    v16bf a = load_a_bf(hrow, kc, half);
#pragma unroll
    for (int s = 0; s < 4; ++s) {
      v16bf b = load_b_bf(Whhb + (size_t)(tileN + s * 16 + r) * H_, kc, half);
      acc[s] = __builtin_amdgcn_wmma_f32_16x16x32_bf16(false, a, false, b,
                                                       (short)0, acc[s], false, false);
    }
  }
  // D layout: lane holds D[m][n], m = tileM + half*8 + i, n = tileN + s*16 + r
#pragma unroll
  for (int s = 0; s < 4; ++s) {
    int n = tileN + s * 16 + r;
    float bias = bih[n] + bhh[n];
#pragma unroll
    for (int i = 0; i < 8; ++i) {
      int m = tileM + half * 8 + i;
      G[(size_t)m * G4H + n] = acc[s][i] + bias;
    }
  }
}

// ---------------------------------------------------------------------------
// LSTM pointwise: gate order i,f,g,o ; fp32 cell state, bf16 hidden output
__global__ void lstm_pointwise_kernel(const float* __restrict__ G,
                                      const float* __restrict__ Cprev,
                                      float* __restrict__ Cnew,
                                      unsigned short* __restrict__ HnewBf) {
  int idx = blockIdx.x * blockDim.x + threadIdx.x;  // B*H = 65536
  int b = idx >> 9, k = idx & (H_ - 1);
  const float* g = G + (size_t)b * G4H;
  float gi = g[k];
  float gf = g[H_ + k];
  float gg = g[2 * H_ + k];
  float go = g[3 * H_ + k];
  float c = sigm(gf) * Cprev[idx] + sigm(gi) * tanhf(gg);
  Cnew[idx] = c;
  HnewBf[idx] = f2bf(sigm(go) * tanhf(c));
}

// ---------------------------------------------------------------------------
// Logits GEMM + gumbel argmax.  wave tile = 32x64 (2 M-tiles, B-frag reuse) ;
// grid = (125, 4) -> 2000 waves.  Streaming logits/gumbel use NT hints so the
// L2-resident bf16 W_fc (32.8MB of 192MB L2) is not evicted.
__global__ void __launch_bounds__(128)
logits_gemm_kernel(const unsigned short* __restrict__ Hb,
                   const unsigned short* __restrict__ Wfcb,
                   const float* __restrict__ bfc, const float* __restrict__ gumb,
                   float* __restrict__ outp, unsigned long long* __restrict__ packed) {
  int wave = threadIdx.x >> 5;
  int lane = threadIdx.x & 31;
  int half = lane >> 4, r = lane & 15;
  int tileN = (blockIdx.x * 4 + wave) * 64;
  int tileM = blockIdx.y * 32;

  const unsigned short* h0row = Hb + (size_t)(tileM + r) * H_;
  const unsigned short* h1row = Hb + (size_t)(tileM + 16 + r) * H_;

  v8f acc[2][4] = {{v8f{}, v8f{}, v8f{}, v8f{}}, {v8f{}, v8f{}, v8f{}, v8f{}}};

  for (int kc = 0; kc < H_; kc += 32) {
    v16bf a0 = load_a_bf(h0row, kc, half);
    v16bf a1 = load_a_bf(h1row, kc, half);
#pragma unroll
    for (int s = 0; s < 4; ++s) {
      v16bf b = load_b_bf(Wfcb + (size_t)(tileN + s * 16 + r) * H_, kc, half);
      acc[0][s] = __builtin_amdgcn_wmma_f32_16x16x32_bf16(false, a0, false, b,
                                                          (short)0, acc[0][s], false, false);
      acc[1][s] = __builtin_amdgcn_wmma_f32_16x16x32_bf16(false, a1, false, b,
                                                          (short)0, acc[1][s], false, false);
    }
  }

  unsigned long long best[2][8];
#pragma unroll
  for (int mt = 0; mt < 2; ++mt)
#pragma unroll
    for (int i = 0; i < 8; ++i) best[mt][i] = 0ull;

#pragma unroll
  for (int s = 0; s < 4; ++s) {
    int n = tileN + s * 16 + r;
    float bias = bfc[n];
#pragma unroll
    for (int mt = 0; mt < 2; ++mt) {
#pragma unroll
      for (int i = 0; i < 8; ++i) {
        int m = tileM + mt * 16 + half * 8 + i;
        float lg = acc[mt][s][i] + bias;
        __builtin_nontemporal_store(lg, &outp[(size_t)m * ((size_t)S_ * V_) + n]);
        float key = lg + __builtin_nontemporal_load(&gumb[(size_t)m * V_ + n]);
        unsigned long long pk =
            ((unsigned long long)ordf(key) << 32) | (unsigned int)(~(unsigned int)n);
        if (pk > best[mt][i]) best[mt][i] = pk;
      }
    }
  }
  // reduce across the 16 lanes of this half (masks 1,2,4,8 stay within half)
#pragma unroll
  for (int msk = 1; msk <= 8; msk <<= 1) {
#pragma unroll
    for (int mt = 0; mt < 2; ++mt)
#pragma unroll
      for (int i = 0; i < 8; ++i) {
        unsigned long long o = shfl_xor_u64(best[mt][i], msk);
        if (o > best[mt][i]) best[mt][i] = o;
      }
  }
  if (r == 0) {
#pragma unroll
    for (int mt = 0; mt < 2; ++mt)
#pragma unroll
      for (int i = 0; i < 8; ++i)
        atomicMax(&packed[tileM + mt * 16 + half * 8 + i], best[mt][i]);
  }
}

// ---------------------------------------------------------------------------
__global__ void finalize_step_kernel(unsigned long long* __restrict__ packed,
                                     int* __restrict__ tok,
                                     float* __restrict__ samples_t /* +b*S */) {
  int b = threadIdx.x;
  unsigned long long pk = packed[b];
  unsigned int n = ~(unsigned int)(pk & 0xFFFFFFFFull);
  tok[b] = (int)n;
  samples_t[(size_t)b * S_] = (float)n;
  packed[b] = 0ull;   // reset for next step
}

// ---------------------------------------------------------------------------
extern "C" void kernel_launch(void* const* d_in, const int* in_sizes, int n_in,
                              void* d_out, int out_size, void* d_ws, size_t ws_size,
                              hipStream_t stream) {
  (void)in_sizes; (void)n_in; (void)out_size; (void)ws_size;

  const float* noise = (const float*)d_in[0];
  const float* emb   = (const float*)d_in[1];
  const float* W_lat = (const float*)d_in[2];
  const float* b_lat = (const float*)d_in[3];
  const float* W_ih0 = (const float*)d_in[4];
  const float* W_hh0 = (const float*)d_in[5];
  const float* b_ih0 = (const float*)d_in[6];
  const float* b_hh0 = (const float*)d_in[7];
  const float* W_ihr = (const float*)d_in[8];   // (NL-1, 4H, H)
  const float* W_hhr = (const float*)d_in[9];
  const float* b_ihr = (const float*)d_in[10];  // (NL-1, 4H)
  const float* b_hhr = (const float*)d_in[11];
  const float* W_fc  = (const float*)d_in[12];
  const float* b_fc  = (const float*)d_in[13];
  const float* gumbel = (const float*)d_in[14]; // (S, B, V)

  // ---- workspace carve-out (~64 MB, 256B-aligned chunks) ----
  char* cur = (char*)d_ws;
  auto carve = [&](size_t bytes) -> void* {
    void* p = (void*)cur;
    cur += (bytes + 255) & ~(size_t)255;
    return p;
  };
  const size_t STATE = (size_t)NL_ * B_ * H_;                 // 196608
  unsigned long long* packed = (unsigned long long*)carve(B_ * 8);
  int*   tok  = (int*)carve(B_ * 4);
  float* cA   = (float*)carve(STATE * 4);
  float* cB   = (float*)carve(STATE * 4);
  float* G    = (float*)carve((size_t)B_ * G4H * 4);
  unsigned short* hbfA = (unsigned short*)carve(STATE * 2);
  unsigned short* hbfB = (unsigned short*)carve(STATE * 2);
  unsigned short* embb = (unsigned short*)carve((size_t)V_ * E_ * 2);
  unsigned short* wih0b = (unsigned short*)carve((size_t)G4H * E_ * 2);
  unsigned short* whh0b = (unsigned short*)carve((size_t)G4H * H_ * 2);
  unsigned short* wihrb = (unsigned short*)carve((size_t)(NL_ - 1) * G4H * H_ * 2);
  unsigned short* whhrb = (unsigned short*)carve((size_t)(NL_ - 1) * G4H * H_ * 2);
  unsigned short* wfcb  = (unsigned short*)carve((size_t)V_ * H_ * 2);

  float* logits_base  = (float*)d_out;                        // (B,S,V)
  float* samples_base = logits_base + (size_t)B_ * S_ * V_;   // (B,S) as float

  // ---- one-time weight/embedding bf16 conversion ----
  auto cvt = [&](const float* s, unsigned short* d, size_t n) {
    int n2 = (int)(n / 2);
    cvt_bf16_kernel<<<(n2 + 255) / 256, 256, 0, stream>>>(s, (unsigned int*)d, n2);
  };
  cvt(emb,   embb,  (size_t)V_ * E_);
  cvt(W_ih0, wih0b, (size_t)G4H * E_);
  cvt(W_hh0, whh0b, (size_t)G4H * H_);
  cvt(W_ihr, wihrb, (size_t)(NL_ - 1) * G4H * H_);
  cvt(W_hhr, whhrb, (size_t)(NL_ - 1) * G4H * H_);
  cvt(W_fc,  wfcb,  (size_t)V_ * H_);

  init_h0_kernel<<<(NL_ * B_ * H_ + 255) / 256, 256, 0, stream>>>(noise, W_lat, b_lat, hbfA, cA);
  init_misc_kernel<<<1, B_, 0, stream>>>(tok, packed);

  unsigned short* hs = hbfA; unsigned short* hn = hbfB;
  float* cs = cA; float* cn = cB;
  const size_t LB = (size_t)B_ * H_;     // per-layer state block (65536)

  for (int t = 0; t < S_; ++t) {
    // layer 0 (input = bf16 embedding gather, K=E)
    gates_gemm_kernel<<<dim3(G4H / 256, B_ / 16), 128, 0, stream>>>(
        embb, tok, hs + 0 * LB, wih0b, whh0b, b_ih0, b_hh0, G, E_);
    lstm_pointwise_kernel<<<(B_ * H_) / 256, 256, 0, stream>>>(
        G, cs + 0 * LB, cn + 0 * LB, hn + 0 * LB);

    for (int l = 1; l < NL_; ++l) {
      gates_gemm_kernel<<<dim3(G4H / 256, B_ / 16), 128, 0, stream>>>(
          hn + (size_t)(l - 1) * LB, nullptr, hs + (size_t)l * LB,
          wihrb + (size_t)(l - 1) * G4H * H_, whhrb + (size_t)(l - 1) * G4H * H_,
          b_ihr + (size_t)(l - 1) * G4H, b_hhr + (size_t)(l - 1) * G4H, G, H_);
      lstm_pointwise_kernel<<<(B_ * H_) / 256, 256, 0, stream>>>(
          G, cs + (size_t)l * LB, cn + (size_t)l * LB, hn + (size_t)l * LB);
    }

    logits_gemm_kernel<<<dim3(V_ / 256, B_ / 32), 128, 0, stream>>>(
        hn + (size_t)(NL_ - 1) * LB, wfcb, b_fc,
        gumbel + (size_t)t * B_ * V_, logits_base + (size_t)t * V_, packed);

    finalize_step_kernel<<<1, B_, 0, stream>>>(packed, tok, samples_base + t);

    // ping-pong states for next step
    unsigned short* th = hs; hs = hn; hn = th;
    float* tc = cs; cs = cn; cn = tc;
  }
}
